// Model_73624329388058
// MI455X (gfx1250) — matrix operand order repaired and checked by reading, not compiled, not run on hardware
//
#include <hip/hip_runtime.h>

typedef _Float16 v8h  __attribute__((ext_vector_type(8)));
typedef _Float16 v16h __attribute__((ext_vector_type(16)));
typedef float    v8f  __attribute__((ext_vector_type(8)));

// gfx1250 has native V_TANH_F32 (TRANS op). Use it if the toolchain exposes it.
#if defined(__has_builtin)
#if __has_builtin(__builtin_amdgcn_tanhf)
#define LSTM_HAS_NATIVE_TANH 1
#endif
#endif

namespace {
constexpr int kH = 32;        // hidden size
constexpr int kT = 15;        // sequence length
constexpr int kRows = 16;     // batch rows per wave (WMMA M)
constexpr int kWaves = 8;     // waves per block
constexpr int kThreads = kWaves * 32;

__device__ __forceinline__ float tanh_f(float x) {
#ifdef LSTM_HAS_NATIVE_TANH
    return __builtin_amdgcn_tanhf(x);
#else
    return 2.0f / (1.0f + __expf(-2.0f * x)) - 1.0f;
#endif
}
__device__ __forceinline__ float sigmoid_f(float x) {
#ifdef LSTM_HAS_NATIVE_TANH
    // sigmoid(x) = 0.5 * tanh(x/2) + 0.5  -> mul + v_tanh_f32 + fma
    return fmaf(0.5f, __builtin_amdgcn_tanhf(0.5f * x), 0.5f);
#else
    return 1.0f / (1.0f + __expf(-x));
#endif
}
} // namespace

__global__ __launch_bounds__(kThreads)
void lstm_wmma_kernel(const float* __restrict__ x,
                      const float* __restrict__ W_ih,
                      const float* __restrict__ W_hh,
                      const float* __restrict__ b_ih,
                      const float* __restrict__ b_hh,
                      const float* __restrict__ W3,
                      const float* __restrict__ b3,
                      const float* __restrict__ W4,
                      const float* __restrict__ b4,
                      float* __restrict__ out,
                      int batch)
{
    // wave-private LDS staging (no cross-wave sharing -> no barriers)
    __shared__ __align__(16) _Float16 hbuf_s[kWaves][kRows * kH];  // h tile, f16 row-major
    __shared__ __align__(16) float    xbuf_s[kWaves][kT * kRows];  // x as [t][row]
    __shared__ __align__(16) float    obuf_s[kWaves][kRows * kH];  // relu layer out, f32

    const int lane = threadIdx.x & 31;
    const int wv   = threadIdx.x >> 5;
    _Float16* hbuf = hbuf_s[wv];
    float*    xbuf = xbuf_s[wv];
    float*    obuf = obuf_s[wv];

    const long m0 = (long)(blockIdx.x * kWaves + wv) * kRows;

    const int col = lane & 15;   // B/D tile column (N); also A row (M)
    const int hi  = lane >> 4;   // lane half: selects K range in A/B layouts

    // ---- constant WMMA B operands: W_hh^T tiles [K=32 x N=16], 8 tiles cover 128 gates.
    // B as loaded here (contiguous k, 32x16): lane n<16 -> N=n, VGPR j holds K=2j,2j+1; lanes 16-31: K+=16.
    // B[k, 16n+c] = W_hh[16n+c, k]  (row 16n+col of W_hh, 16 consecutive k per lane half)
    v16h Bh[8];
#pragma unroll
    for (int n = 0; n < 8; ++n) {
        const float* wrow = W_hh + (n * 16 + col) * kH + hi * 16;
        v16h b;
#pragma unroll
        for (int i = 0; i < 16; ++i) b[i] = (_Float16)wrow[i];
        Bh[n] = b;
    }
    // W3^T tiles (2 tiles of N=16) + b3 per-lane bias
    v16h B3[2];
    float bias3[2];
#pragma unroll
    for (int n = 0; n < 2; ++n) {
        const float* wrow = W3 + (n * 16 + col) * kH + hi * 16;
        v16h b;
#pragma unroll
        for (int i = 0; i < 16; ++i) b[i] = (_Float16)wrow[i];
        B3[n] = b;
        bias3[n] = b3[n * 16 + col];
    }
    // per-lane gate bias (b_ih+b_hh) and input weight W_ih, indexed by gate column N
    float bsum[8], wih[8];
#pragma unroll
    for (int n = 0; n < 8; ++n) {
        const int j = n * 16 + col;
        bsum[n] = b_ih[j] + b_hh[j];
        wih[n]  = W_ih[j];
    }

    // ---- stage x tile into LDS as [t][row] (contiguous per-step broadcast reads)
    for (int idx = lane; idx < kT * kRows; idx += 32) {
        const int r = idx / kT;
        const int t = idx - r * kT;
        long m = m0 + r;
        if (m >= batch) m = batch - 1;   // clamp tail (outputs guarded below)
        xbuf[t * kRows + r] = x[m * kT + t];
    }
    // ---- h0 = 0
    for (int idx = lane; idx < kRows * kH; idx += 32) hbuf[idx] = (_Float16)0.0f;
    __builtin_amdgcn_fence(__ATOMIC_ACQ_REL, "wavefront");

    v8f cst0 = {}, cst1 = {};   // cell state in D-tile layout (hidden 0-15 / 16-31)

    // A-layout chunk pointers (16-bit A 16x32):
    // lanes<16: elems0-7 <- h[r,0..7],  elems8-15 <- h[r,8..15]
    // lanes>=16: elems0-7 <- h[r,16..23], elems8-15 <- h[r,24..31]
    const _Float16* aptr0 = hbuf + col * kH + hi * 16;
    const _Float16* aptr1 = aptr0 + 8;

    for (int t = 0; t < kT; ++t) {
        // A operand: rebuild h in A layout from LDS (2x ds_load_b128 per lane)
        v8h alo = *(const v8h*)aptr0;
        v8h ahi = *(const v8h*)aptr1;
        v16h A;
#pragma unroll
        for (int i = 0; i < 8; ++i) { A[i] = alo[i]; A[8 + i] = ahi[i]; }

        // x values for this lane's 8 D-rows (m = v + hi*8)
        const float* xp = xbuf + t * kRows + hi * 8;
        float xv[8];
#pragma unroll
        for (int v = 0; v < 8; ++v) xv[v] = xp[v];

        // gates[16x128] = h @ W_hh^T + (b + x*W_ih)  -> 8 WMMAs, bias folded into C
        v8f acc[8];
#pragma unroll
        for (int n = 0; n < 8; ++n) {
            v8f c;
#pragma unroll
            for (int v = 0; v < 8; ++v) c[v] = fmaf(xv[v], wih[n], bsum[n]);
            acc[n] = __builtin_amdgcn_wmma_f32_16x16x32_f16(
                false, A, false, Bh[n], (short)0, c, false, false);
        }

        // torch gate order i,f,g,o -> tiles {0,1},{2,3},{4,5},{6,7}
#pragma unroll
        for (int hf = 0; hf < 2; ++hf) {
            v8f& cs = hf ? cst1 : cst0;
#pragma unroll
            for (int v = 0; v < 8; ++v) {
                const float ig = sigmoid_f(acc[0 + hf][v]);
                const float fg = sigmoid_f(acc[2 + hf][v]);
                const float gg = tanh_f  (acc[4 + hf][v]);
                const float og = sigmoid_f(acc[6 + hf][v]);
                const float cn = fmaf(fg, cs[v], ig * gg);
                cs[v] = cn;
                const float hn = og * tanh_f(cn);
                // store h element (row m, hidden k) to LDS row-major f16
                hbuf[(v + hi * 8) * kH + hf * 16 + col] = (_Float16)hn;
            }
        }
        __builtin_amdgcn_fence(__ATOMIC_ACQ_REL, "wavefront");
    }

    // ---- out = relu(h @ W3^T + b3): 2 WMMAs, result to LDS (f32)
    {
        v8h alo = *(const v8h*)aptr0;
        v8h ahi = *(const v8h*)aptr1;
        v16h A;
#pragma unroll
        for (int i = 0; i < 8; ++i) { A[i] = alo[i]; A[8 + i] = ahi[i]; }
#pragma unroll
        for (int n = 0; n < 2; ++n) {
            v8f c;
#pragma unroll
            for (int v = 0; v < 8; ++v) c[v] = bias3[n];
            v8f o3 = __builtin_amdgcn_wmma_f32_16x16x32_f16(
                false, A, false, B3[n], (short)0, c, false, false);
#pragma unroll
            for (int v = 0; v < 8; ++v)
                obuf[(v + hi * 8) * kH + n * 16 + col] = fmaxf(o3[v], 0.0f);
        }
    }
    __builtin_amdgcn_fence(__ATOMIC_ACQ_REL, "wavefront");

    // ---- y = sigmoid(out @ W4^T + b4): one batch row per lane 0..15
    if (lane < 16 && (m0 + lane) < batch) {
        float acc = b4[0];
        const float* row = obuf + lane * kH;
#pragma unroll
        for (int k = 0; k < kH; ++k) acc = fmaf(row[k], W4[k], acc);
        out[m0 + lane] = sigmoid_f(acc);
    }
}

extern "C" void kernel_launch(void* const* d_in, const int* in_sizes, int n_in,
                              void* d_out, int out_size, void* d_ws, size_t ws_size,
                              hipStream_t stream) {
    const float* x    = (const float*)d_in[0];
    const float* W_ih = (const float*)d_in[1];
    const float* W_hh = (const float*)d_in[2];
    const float* b_ih = (const float*)d_in[3];
    const float* b_hh = (const float*)d_in[4];
    const float* W3   = (const float*)d_in[5];
    const float* b3   = (const float*)d_in[6];
    const float* W4   = (const float*)d_in[7];
    const float* b4   = (const float*)d_in[8];
    float* out = (float*)d_out;

    const int batch  = in_sizes[0] / kT;                       // x has B*T*1 elements
    const int waves  = (batch + kRows - 1) / kRows;
    const int blocks = (waves + kWaves - 1) / kWaves;

    lstm_wmma_kernel<<<dim3(blocks), dim3(kThreads), 0, stream>>>(
        x, W_ih, W_hh, b_ih, b_hh, W3, b3, W4, b4, out, batch);
}
